// Linear_28123445854372
// MI455X (gfx1250) — 8-bit matrix operand order repaired and checked by reading, compiled, not run on hardware
//
#include <hip/hip_runtime.h>
#include <hip/hip_bf16.h>

#define M_TOTAL 8192
#define K_TOTAL 7168
#define N_TOTAL 7168
#define QB      128
#define KB      (K_TOTAL / QB)
#define NBLK    (N_TOTAL / QB)

typedef __attribute__((ext_vector_type(16))) int   v16i;
typedef __attribute__((ext_vector_type(8)))  float v8f;

__device__ __forceinline__ void async_b64_x8(unsigned ldsOff,
                                             const unsigned char* g) {
    asm volatile(
        "global_load_async_to_lds_b64 %0, %1, off\n\t"
        "global_load_async_to_lds_b64 %0, %1, off offset:8\n\t"
        "global_load_async_to_lds_b64 %0, %1, off offset:16\n\t"
        "global_load_async_to_lds_b64 %0, %1, off offset:24\n\t"
        "global_load_async_to_lds_b64 %0, %1, off offset:32\n\t"
        "global_load_async_to_lds_b64 %0, %1, off offset:40\n\t"
        "global_load_async_to_lds_b64 %0, %1, off offset:48\n\t"
        "global_load_async_to_lds_b64 %0, %1, off offset:56"
        :: "v"(ldsOff), "v"(g) : "memory");
}
__device__ __forceinline__ void async_b128_x4(unsigned ldsOff,
                                              const unsigned char* g) {
    asm volatile(
        "global_load_async_to_lds_b128 %0, %1, off\n\t"
        "global_load_async_to_lds_b128 %0, %1, off offset:16\n\t"
        "global_load_async_to_lds_b128 %0, %1, off offset:32\n\t"
        "global_load_async_to_lds_b128 %0, %1, off offset:48"
        :: "v"(ldsOff), "v"(g) : "memory");
}
__device__ __forceinline__ void wait_async0() {
    asm volatile("s_wait_asynccnt 0" ::: "memory");
}
__device__ __forceinline__ unsigned lds_off(const void* p) {
    return (unsigned)(unsigned long long)p;
}

__device__ __forceinline__ unsigned char f32_to_e4m3(float f) {
    unsigned char sgn = (__float_as_uint(f) >> 31) ? 0x80 : 0x00;
    float a = fabsf(f);
    if (!(a == a)) return sgn | 0x7F;
    if (a >= 464.0f) return sgn | 0x7E;
    if (a == 0.0f) return sgn;
    int e;
    frexpf(a, &e);
    int se = e - 4;
    if (se < -9) se = -9;
    float step = ldexpf(1.0f, se);
    float q = rintf(a / step) * step;
    if (q == 0.0f) return sgn;
    int qe; float qm = frexpf(q, &qe);
    if (qe - 1 < -6) {
        return sgn | (unsigned char)(q * 512.0f);
    }
    int E = qe - 1 + 7;
    int Mant = (int)(qm * 16.0f) - 8;
    return sgn | (unsigned char)((E << 3) | Mant);
}

__global__ __launch_bounds__(256)
void act_quant_kernel(const float* __restrict__ x,
                      unsigned char* __restrict__ xq,
                      float* __restrict__ xs) {
    __shared__ float red[256];
    const int tid = threadIdx.x;
    const size_t e = (size_t)blockIdx.x * 256 + tid;
    const float v = x[e];
    red[tid] = fabsf(v);
    __syncthreads();
    const int g = tid & 127;
    #pragma unroll
    for (int off = 64; off >= 1; off >>= 1) {
        if (g < off) red[tid] = fmaxf(red[tid], red[tid + off]);
        __syncthreads();
    }
    const float amax = red[tid - g];
    const float s = fmaxf(amax * (1.0f / 448.0f), 1e-30f);
    xq[e] = f32_to_e4m3(v / s);
    if (g == 0) xs[e >> 7] = s;
}

__global__ __launch_bounds__(256)
void w_quant_kernel(const float* __restrict__ w,
                    unsigned char* __restrict__ wq) {
    const size_t i = ((size_t)blockIdx.x * 256 + threadIdx.x) * 4;
    const float4 v = *(const float4*)(w + i);
    uchar4 o;
    o.x = f32_to_e4m3(v.x);
    o.y = f32_to_e4m3(v.y);
    o.z = f32_to_e4m3(v.z);
    o.w = f32_to_e4m3(v.w);
    *(uchar4*)(wq + i) = o;
}

#define LDA 136
#define LDB 144

__global__ __launch_bounds__(256)
void fp8_gemm_kernel(const unsigned char* __restrict__ Aq,
                     const float*         __restrict__ Asc,
                     const unsigned char* __restrict__ Bq,
                     const float*         __restrict__ Bsc,
                     float*               __restrict__ Out)
{
    __shared__ unsigned char Alds[2][128 * LDA];
    __shared__ unsigned char Blds[2][128 * LDB];
    __shared__ float         XsLds[2][128];

    const int tid   = threadIdx.x;
    const int lane  = tid & 31;
    const int wid   = tid >> 5;
    const int wm    = wid >> 1;
    const int wn    = wid & 1;
    const int laneM = lane & 15;
    const int h     = lane >> 4;

    const int mBase = blockIdx.y * 128;
    const int nBase = blockIdx.x * 128;

    float acc[2][4][8];
    #pragma unroll
    for (int mt = 0; mt < 2; ++mt)
        #pragma unroll
        for (int nt = 0; nt < 4; ++nt)
            #pragma unroll
            for (int j = 0; j < 8; ++j) acc[mt][nt][j] = 0.0f;

    const int srow = tid >> 1;
    const int scol = (tid & 1) * 64;

    const unsigned char* aptr = Aq + (size_t)(mBase + srow) * K_TOTAL + scol;
    const unsigned char* bptr = Bq + (size_t)(nBase + srow) * K_TOTAL + scol;

    unsigned laoff[2], lboff[2];
    #pragma unroll
    for (int b = 0; b < 2; ++b) {
        laoff[b] = lds_off(&Alds[b][srow * LDA + scol]);
        lboff[b] = lds_off(&Blds[b][srow * LDB + scol]);
    }

    async_b64_x8(laoff[0], aptr);
    async_b128_x4(lboff[0], bptr);
    {
        float s0 = 0.0f;
        if (tid < 128) s0 = Asc[(size_t)(mBase + tid) * KB + 0];
        wait_async0();
        if (tid < 128) XsLds[0][tid] = s0;
    }
    __syncthreads();

    for (int kb = 0; kb < KB; ++kb) {
        const int cur = kb & 1;
        const int nxt = cur ^ 1;

        float snext = 0.0f;
        if (kb + 1 < KB) {
            const int k1 = (kb + 1) * QB;
            async_b64_x8(laoff[nxt], aptr + k1);
            async_b128_x4(lboff[nxt], bptr + k1);
            if (tid < 128) snext = Asc[(size_t)(mBase + tid) * KB + (kb + 1)];
        }
        const float ws = Bsc[blockIdx.x * KB + kb];

        v16i afrag[2];
        #pragma unroll
        for (int mt = 0; mt < 2; ++mt) {
            const unsigned char* ab =
                &Alds[cur][(wm * 32 + mt * 16 + laneM) * LDA + 16 * h];
            #pragma unroll
            for (int j = 0; j < 8; ++j) {
                const unsigned long long d =
                    *(const unsigned long long*)(ab + (j & 1) * 8 + (j >> 1) * 32);
                afrag[mt][2 * j]     = (int)(unsigned int)d;
                afrag[mt][2 * j + 1] = (int)(d >> 32);
            }
        }
        v16i bfrag[4];
        #pragma unroll
        for (int nt = 0; nt < 4; ++nt) {
            const unsigned char* bb =
                &Blds[cur][(wn * 64 + nt * 16 + laneM) * LDB + 16 * h];
            #pragma unroll
            for (int j = 0; j < 4; ++j) {
                const uint4 d = *(const uint4*)(bb + 32 * j);
                bfrag[nt][4 * j]     = (int)d.x;
                bfrag[nt][4 * j + 1] = (int)d.y;
                bfrag[nt][4 * j + 2] = (int)d.z;
                bfrag[nt][4 * j + 3] = (int)d.w;
            }
        }
        float xsf[2][8];
        #pragma unroll
        for (int mt = 0; mt < 2; ++mt) {
            const float* xp = &XsLds[cur][wm * 32 + mt * 16 + 8 * h];
            #pragma unroll
            for (int j = 0; j < 8; ++j) xsf[mt][j] = xp[j] * ws;
        }

        const v8f zeroC = {0.f, 0.f, 0.f, 0.f, 0.f, 0.f, 0.f, 0.f};
        #pragma unroll
        for (int mt = 0; mt < 2; ++mt)
            #pragma unroll
            for (int nt = 0; nt < 4; ++nt) {
                v8f c = __builtin_amdgcn_wmma_f32_16x16x128_fp8_fp8(
                    afrag[mt], bfrag[nt], (short)0, zeroC, false, false);
                #pragma unroll
                for (int j = 0; j < 8; ++j)
                    acc[mt][nt][j] += c[j] * xsf[mt][j];
            }

        if (kb + 1 < KB) {
            wait_async0();
            if (tid < 128) XsLds[nxt][tid] = snext;
            __syncthreads();
        }
    }

    #pragma unroll
    for (int mt = 0; mt < 2; ++mt)
        #pragma unroll
        for (int nt = 0; nt < 4; ++nt) {
            const int col = nBase + wn * 64 + nt * 16 + laneM;
            #pragma unroll
            for (int j = 0; j < 8; ++j) {
                const int row = mBase + wm * 32 + mt * 16 + 8 * h + j;
                Out[(size_t)row * N_TOTAL + col] = acc[mt][nt][j];
            }
        }
}

extern "C" void kernel_launch(void* const* d_in, const int* in_sizes, int n_in,
                              void* d_out, int out_size, void* d_ws, size_t ws_size,
                              hipStream_t stream) {
    const float* x    = (const float*)d_in[0];
    const float* wq   = (const float*)d_in[1];
    const float* wsc  = (const float*)d_in[2];
    float*       out  = (float*)d_out;

    unsigned char* xq8 = (unsigned char*)d_ws;
    size_t off = (size_t)M_TOTAL * K_TOTAL;
    unsigned char* wq8 = xq8 + off;
    off += (size_t)N_TOTAL * K_TOTAL;
    off = (off + 255) & ~(size_t)255;
    float* xsc = (float*)(xq8 + off);

    act_quant_kernel<<<(M_TOTAL * (size_t)K_TOTAL) / 256, 256, 0, stream>>>(
        x, xq8, xsc);

    w_quant_kernel<<<(N_TOTAL * (size_t)K_TOTAL) / 1024, 256, 0, stream>>>(
        wq, wq8);

    dim3 grid(N_TOTAL / 128, M_TOTAL / 128);
    fp8_gemm_kernel<<<grid, 256, 0, stream>>>(xq8, xsc, wq8, wsc, out);
}
